// SecondOrderScorer_44659069943931
// MI455X (gfx1250) — compile-verified, matrix operand order checked by reading, not run on hardware
//
#include <hip/hip_runtime.h>
#include <hip/hip_bf16.h>

// ---------------------------------------------------------------------------
// SecondOrderScorer on MI455X (gfx1250, wave32, v_wmma_f32_16x16x32_f16).
//
// ~45 GFLOP of matmul vs 191 MB output (8us @ 23.3 TB/s) -> compute bound.
// All operands K-contiguous, zero-padded to K=160: each WMMA fragment is two
// aligned 16-byte loads, no bounds checks in any inner loop. Register tiling
// (2x4 in the big GEMM, 3x3 in the scorer) gives fragment reuse so multiple
// v_wmma issue per pair of b128 loads. W_tri f16 copy (65 MB) is L2-resident.
// ---------------------------------------------------------------------------

typedef __attribute__((ext_vector_type(16))) _Float16 v16h;
typedef __attribute__((ext_vector_type(8)))  _Float16 v8h;
typedef __attribute__((ext_vector_type(8)))  float    v8f;

#define B_    2
#define L_    144
#define DIN_  1024
#define DT_   150
#define NOUT_ 4
#define BL_   (B_ * L_)                  // 288
#define DTP_  160                        // DT padded to multiple of 32
#define NWP_  (DTP_ * DTP_)              // 25600 padded (i,j) columns
#define HALF_OUT (B_ * L_ * L_ * L_ * NOUT_)   // 23,887,872 floats per tri

// workspace layout (units: _Float16 elements); all offsets 16B-aligned
#define WS_WP  0                                   // Wproj_h[4][DTP][DIN]
#define WS_P   (WS_WP + 4 * DTP_ * DIN_)           // Ph[4][BL][DTP]
#define WS_WT  (WS_P + 4 * BL_ * DTP_)             // Wth[2][4][DTP][DTP][DTP]
#define WS_W   (WS_WT + 2 * NOUT_ * DTP_ * NWP_)   // wh[2][BL][4][NWP]
// total ~185 MB of d_ws

__device__ __forceinline__ v16h load_frag(const _Float16* rowp, int k0) {
  // lane fragment = row[k0..k0+7] ++ row[k0+16..k0+23]; both 16B aligned
  v8h lo = *(const v8h*)(rowp + k0);
  v8h hi = *(const v8h*)(rowp + k0 + 16);
  return __builtin_shufflevector(lo, hi, 0, 1, 2, 3, 4, 5, 6, 7,
                                 8, 9, 10, 11, 12, 13, 14, 15);
}

__device__ __forceinline__ v8f wmma_f16(v16h a, v16h b, v8f c) {
  return __builtin_amdgcn_wmma_f32_16x16x32_f16(
      false, a, false, b, (short)0, c, false, false);
}

// ---------------------------------------------------------------------------
// Prep A: Wproj_h[p][t][k] = (f16) W_p[k*DT + t], zero for t >= 150.
// ---------------------------------------------------------------------------
__global__ void prep_projw_kernel(const float* __restrict__ W0,
                                  const float* __restrict__ W1,
                                  const float* __restrict__ W2,
                                  const float* __restrict__ W3,
                                  _Float16* __restrict__ Wp) {
  int idx = blockIdx.x * blockDim.x + threadIdx.x;   // 4*160*1024
  int k = idx & (DIN_ - 1);
  int t = (idx >> 10) % DTP_;
  int p = idx / (DIN_ * DTP_);
  if (p >= 4) return;
  const float* W = (p == 0) ? W0 : (p == 1) ? W1 : (p == 2) ? W2 : W3;
  Wp[idx] = (t < DT_) ? (_Float16)W[k * DT_ + t] : (_Float16)0.f;
}

// ---------------------------------------------------------------------------
// Prep B: Wth[tri][o][i][j][k] = (f16) W_tri[tri][o][i][k][j], zero padded.
// ---------------------------------------------------------------------------
__global__ void prep_tri_kernel(const float* __restrict__ Wt1,
                                const float* __restrict__ Wt2,
                                _Float16* __restrict__ Wth) {
  long long idx = (long long)blockIdx.x * blockDim.x + threadIdx.x;
  const long long total = 2LL * NOUT_ * DTP_ * DTP_ * DTP_;  // 32,768,000
  if (idx >= total) return;
  long long t = idx;
  int k = (int)(t % DTP_); t /= DTP_;
  int j = (int)(t % DTP_); t /= DTP_;
  int i = (int)(t % DTP_); t /= DTP_;
  int o = (int)(t % NOUT_); t /= NOUT_;
  int tri = (int)t;
  _Float16 v = (_Float16)0.f;
  if (i < DT_ && j < DT_ && k < DT_) {
    const float* Wsrc = tri ? Wt2 : Wt1;
    v = (_Float16)Wsrc[((o * DT_ + i) * DT_ + k) * DT_ + j];
  }
  Wth[idx] = v;
}

// ---------------------------------------------------------------------------
// Kernel 1: projections  Ph[p][bz][t] = x[bz,:] @ W_p + b_p  (f16, t padded).
// tiles: p(4) x mt(18) x nt(10); one wave per 16x16 tile; K = 1024.
// ---------------------------------------------------------------------------
__global__ __launch_bounds__(256)
void proj_kernel(const float* __restrict__ x,
                 const float* __restrict__ bv0, const float* __restrict__ bv1,
                 const float* __restrict__ bv2, const float* __restrict__ bv3,
                 const _Float16* __restrict__ Wp,
                 _Float16* __restrict__ Ph) {
  int gid  = blockIdx.x * blockDim.x + threadIdx.x;
  int wid  = gid >> 5;
  int lane = threadIdx.x & 31;
  if (wid >= 720) return;
  int p   = wid / 180;
  int rem = wid % 180;
  int mt = rem / 10, nt = rem % 10;
  const float* bv = (p == 0) ? bv0 : (p == 1) ? bv1 : (p == 2) ? bv2 : bv3;
  int row = mt * 16 + (lane & 15);          // bz < 288
  int col = nt * 16 + (lane & 15);          // t  < 160 (padded)
  int koff = (lane & 16) >> 1;              // 8 for upper half-wave
  const float*    Arow = x + row * DIN_;
  const _Float16* Brow = Wp + (p * DTP_ + col) * DIN_;
  v8f acc = {};
  for (int kb = 0; kb < DIN_; kb += 32) {
    int k0 = kb + koff;
    float4 a0 = *(const float4*)(Arow + k0);
    float4 a1 = *(const float4*)(Arow + k0 + 4);
    float4 a2 = *(const float4*)(Arow + k0 + 16);
    float4 a3 = *(const float4*)(Arow + k0 + 20);
    v16h a;
    a[0] = (_Float16)a0.x;  a[1] = (_Float16)a0.y;
    a[2] = (_Float16)a0.z;  a[3] = (_Float16)a0.w;
    a[4] = (_Float16)a1.x;  a[5] = (_Float16)a1.y;
    a[6] = (_Float16)a1.z;  a[7] = (_Float16)a1.w;
    a[8] = (_Float16)a2.x;  a[9] = (_Float16)a2.y;
    a[10] = (_Float16)a2.z; a[11] = (_Float16)a2.w;
    a[12] = (_Float16)a3.x; a[13] = (_Float16)a3.y;
    a[14] = (_Float16)a3.z; a[15] = (_Float16)a3.w;
    v16h b = load_frag(Brow, k0);
    acc = wmma_f16(a, b, acc);
  }
  float bias = (col < DT_) ? bv[col] : 0.f;   // padded cols: acc==0, bias==0
#pragma unroll
  for (int r = 0; r < 8; ++r) {
    int orow = mt * 16 + r + koff;
    Ph[(p * BL_ + orow) * DTP_ + col] = (_Float16)(acc[r] + bias);
  }
}

// ---------------------------------------------------------------------------
// Kernel 2: big GEMM  wh[tri][bz][o][c'] = sh[bz,:] @ Wth[tri][o][c',:]
// M=288, K=160, N=25600. Register-tiled 32x64 per wave (2 M x 4 N subtiles):
// per K step, 6 fragment loads feed 8 v_wmma (fragment reuse both ways).
// tiles: tri(2) x o(4) x mg(9) x ng(400) = 28,800 waves -> 3,600 blocks.
// ---------------------------------------------------------------------------
__global__ __launch_bounds__(256)
void wbuild_kernel(const _Float16* __restrict__ Ph,
                   const _Float16* __restrict__ Wth,
                   _Float16* __restrict__ wh) {
  int gid  = blockIdx.x * blockDim.x + threadIdx.x;
  int wid  = gid >> 5;
  int lane = threadIdx.x & 31;
  const int NG = NWP_ / 64;                 // 400 groups of 4 N-tiles
  const int total = 2 * NOUT_ * 9 * NG;     // 28,800 wave-tasks
  if (wid >= total) return;
  int ng = wid % NG; int t2 = wid / NG;
  int mg = t2 % 9;   t2 /= 9;
  int o  = t2 % NOUT_; int tri = t2 / NOUT_;
  int koff = (lane & 16) >> 1;
  int row0 = mg * 32 + (lane & 15);         // and row0+16
  const _Float16* Arow0 = Ph + row0 * DTP_;            // sh = projection 0
  const _Float16* Arow1 = Arow0 + 16 * DTP_;
  const _Float16* Bbase =
      Wth + ((long long)(tri * NOUT_ + o) * NWP_ + ng * 64 + (lane & 15)) * DTP_;
  v8f acc[2][4] = {};
#pragma unroll
  for (int kb = 0; kb < DTP_; kb += 32) {
    int k0 = kb + koff;
    v16h a0 = load_frag(Arow0, k0);
    v16h a1 = load_frag(Arow1, k0);
#pragma unroll
    for (int s = 0; s < 4; ++s) {
      v16h b = load_frag(Bbase + s * 16 * DTP_, k0);
      acc[0][s] = wmma_f16(a0, b, acc[0][s]);
      acc[1][s] = wmma_f16(a1, b, acc[1][s]);
    }
  }
#pragma unroll
  for (int m = 0; m < 2; ++m) {
#pragma unroll
    for (int s = 0; s < 4; ++s) {
      int col = ng * 64 + s * 16 + (lane & 15);
#pragma unroll
      for (int r = 0; r < 8; ++r) {
        int bz = mg * 32 + m * 16 + r + koff;
        wh[(((long long)(tri * BL_ + bz)) * NOUT_ + o) * NWP_ + col] =
            (_Float16)acc[m][s][r];
      }
    }
  }
}

// ---------------------------------------------------------------------------
// Kernel 3: per (tri, b, z, o) workgroup:
//   T[y][i] = sum_j M[i][j]*Y[y][j]  -> LDS (y-major), 2x3 register tiles
//   S[x][y] = sum_i X[x][i]*T[y][i]  -> d_out, 3x3 register tiles
// tri1: X=st, Y=oh, S[x][y] -> out[b,z,x,y,o]
// tri2: X=ot, Y=st, S[X][Y] -> out2[b,z,Y,X,o]  (the (0,1,3,2,4) transpose)
// ---------------------------------------------------------------------------
#define TL_STRIDE 168   // f16; 336 B rows, 16B aligned, destaggers banks
__global__ __launch_bounds__(256)
void score_kernel(const _Float16* __restrict__ Ph,
                  const _Float16* __restrict__ wh,
                  float* __restrict__ out) {
  __shared__ _Float16 T_lds[L_][TL_STRIDE];   // 144 x 168 f16 = 48,384 B
  int bi  = blockIdx.x;
  int o   = bi & 3;
  int bz  = (bi >> 2) % BL_;
  int tri = bi / (BL_ * 4);
  int b = bz / L_, z = bz % L_;
  int lane = threadIdx.x & 31;
  int wv   = threadIdx.x >> 5;   // 0..7
  int koff = (lane & 16) >> 1;

  const _Float16* M = wh + (((long long)(tri * BL_ + bz)) * NOUT_ + o) * NWP_;
  int yproj = (tri == 0) ? 2 : 1;   // oh : st
  int xproj = (tri == 0) ? 1 : 3;   // st : ot
  const _Float16* Y = Ph + (yproj * BL_ + b * L_) * DTP_;
  const _Float16* X = Ph + (xproj * BL_ + b * L_) * DTP_;

  // Phase A: 5 x 3 tasks (each 2 i-tiles x 3 y-tiles) over 8 waves.
  for (int t = wv; t < 15; t += 8) {
    int ig = t / 3, yg = t % 3;
    const _Float16* Arow0 = M + (ig * 32 + (lane & 15)) * DTP_;
    const _Float16* Arow1 = Arow0 + 16 * DTP_;
    const _Float16* Brow0 = Y + (yg * 48 + (lane & 15)) * DTP_;
    v8f acc[2][3] = {};
#pragma unroll
    for (int kb = 0; kb < DTP_; kb += 32) {
      int k0 = kb + koff;
      v16h a0 = load_frag(Arow0, k0);
      v16h a1 = load_frag(Arow1, k0);
#pragma unroll
      for (int s = 0; s < 3; ++s) {
        v16h bf = load_frag(Brow0 + s * 16 * DTP_, k0);
        acc[0][s] = wmma_f16(a0, bf, acc[0][s]);
        acc[1][s] = wmma_f16(a1, bf, acc[1][s]);
      }
    }
#pragma unroll
    for (int m = 0; m < 2; ++m) {
#pragma unroll
      for (int s = 0; s < 3; ++s) {
        v8h h;
#pragma unroll
        for (int e = 0; e < 8; ++e) h[e] = (_Float16)acc[m][s][e];
        int ycol = yg * 48 + s * 16 + (lane & 15);
        // 8 consecutive i at fixed y -> one 16B LDS store
        *(v8h*)&T_lds[ycol][ig * 32 + m * 16 + koff] = h;
      }
    }
  }
  __syncthreads();

  // Phase B: 3 x 3 tasks (each 3 x-tiles x 3 y-tiles) over 8 waves.
  for (int t = wv; t < 9; t += 8) {
    int xg = t / 3, yg = t % 3;
    const _Float16* Ar = X + (xg * 48 + (lane & 15)) * DTP_;
    const _Float16* Br = &T_lds[yg * 48 + (lane & 15)][0];
    v8f acc[3][3] = {};
#pragma unroll
    for (int kb = 0; kb < DTP_; kb += 32) {
      int k0 = kb + koff;
      v16h a0 = load_frag(Ar, k0);
      v16h a1 = load_frag(Ar + 16 * DTP_, k0);
      v16h a2 = load_frag(Ar + 32 * DTP_, k0);
#pragma unroll
      for (int s = 0; s < 3; ++s) {
        v16h bf = load_frag(Br + s * 16 * TL_STRIDE, k0);   // 2x ds_load_b128
        acc[0][s] = wmma_f16(a0, bf, acc[0][s]);
        acc[1][s] = wmma_f16(a1, bf, acc[1][s]);
        acc[2][s] = wmma_f16(a2, bf, acc[2][s]);
      }
    }
#pragma unroll
    for (int m = 0; m < 3; ++m) {
#pragma unroll
      for (int s = 0; s < 3; ++s) {
        int ycol = yg * 48 + s * 16 + (lane & 15);
#pragma unroll
        for (int r = 0; r < 8; ++r) {
          int xr = xg * 48 + m * 16 + r + koff;
          long long idx;
          if (tri == 0)
            idx = ((((long long)(b * L_ + z)) * L_ + xr) * L_ + ycol) * NOUT_ + o;
          else
            idx = (long long)HALF_OUT +
                  ((((long long)(b * L_ + z)) * L_ + ycol) * L_ + xr) * NOUT_ + o;
          out[idx] = acc[m][s][r];
        }
      }
    }
  }
}

// ---------------------------------------------------------------------------
// Kernel 4: in-place mask + log_softmax over the 4 contiguous NOUT values.
// Reference zeroes masked entries BEFORE log_softmax (masked -> -log 4).
// ---------------------------------------------------------------------------
__global__ void softmax_kernel(float* __restrict__ out) {
  long long p = (long long)blockIdx.x * blockDim.x + threadIdx.x;
  const long long total = 2LL * B_ * L_ * L_ * L_;
  if (p >= total) return;
  long long t = p / L_;            // drop y
  int x = (int)(t % L_); t /= L_;
  int z = (int)(t % L_);
  float4 v = ((const float4*)out)[p];
  if (z > x) { v.x = 0.f; v.y = 0.f; v.z = 0.f; v.w = 0.f; }
  float m = fmaxf(fmaxf(v.x, v.y), fmaxf(v.z, v.w));
  float s = __expf(v.x - m) + __expf(v.y - m) + __expf(v.z - m) + __expf(v.w - m);
  float lse = m + __logf(s);
  v.x -= lse; v.y -= lse; v.z -= lse; v.w -= lse;
  ((float4*)out)[p] = v;
}

// ---------------------------------------------------------------------------
extern "C" void kernel_launch(void* const* d_in, const int* in_sizes, int n_in,
                              void* d_out, int out_size, void* d_ws, size_t ws_size,
                              hipStream_t stream) {
  const float* x    = (const float*)d_in[0];
  const float* W_sh = (const float*)d_in[1]; const float* b_sh = (const float*)d_in[2];
  const float* W_st = (const float*)d_in[3]; const float* b_st = (const float*)d_in[4];
  const float* W_oh = (const float*)d_in[5]; const float* b_oh = (const float*)d_in[6];
  const float* W_ot = (const float*)d_in[7]; const float* b_ot = (const float*)d_in[8];
  const float* W_tri1 = (const float*)d_in[9];
  const float* W_tri2 = (const float*)d_in[10];
  float*    out = (float*)d_out;
  _Float16* ws  = (_Float16*)d_ws;
  _Float16* Wp  = ws + WS_WP;
  _Float16* Ph  = ws + WS_P;
  _Float16* Wth = ws + WS_WT;
  _Float16* wh  = ws + WS_W;

  // prep: transpose weights into K-contiguous, zero-padded f16
  prep_projw_kernel<<<(4 * DTP_ * DIN_) / 256, 256, 0, stream>>>(
      W_sh, W_st, W_oh, W_ot, Wp);
  prep_tri_kernel<<<(int)((2LL * NOUT_ * DTP_ * DTP_ * DTP_) / 256), 256, 0,
                    stream>>>(W_tri1, W_tri2, Wth);
  // projections: 720 wave-tiles -> 90 blocks x 8 waves
  proj_kernel<<<90, 256, 0, stream>>>(x, b_sh, b_st, b_oh, b_ot, Wp, Ph);
  // M-build GEMM: 28,800 wave-tasks (32x64 each) -> 3,600 blocks
  wbuild_kernel<<<3600, 256, 0, stream>>>(Ph, Wth, wh);
  // per-(tri,b,z,o) scorer: 2304 workgroups
  score_kernel<<<2304, 256, 0, stream>>>(Ph, wh, out);
  // mask + log_softmax: 46,656 blocks x 256
  softmax_kernel<<<46656, 256, 0, stream>>>(out);
}
